// DAttentionBaseline_81870666596609
// MI455X (gfx1250) — matrix operand order repaired and checked by reading, not compiled, not run on hardware
//
#include <hip/hip_runtime.h>
#include <hip/hip_bf16.h>

// ---------------------------------------------------------------------------
// DAttention for MI455X (gfx1250, wave32, WMMA).
// All heavy GEMMs -> v_wmma_f32_16x16x32_bf16 (f32 accumulate), with 2x2
// register blocking per wave (each fragment feeds two WMMAs -> ~8 FLOP/B).
// Attention is fused per 16-row tile with LDS-resident bf16 probabilities.
// ---------------------------------------------------------------------------

typedef __attribute__((ext_vector_type(4)))  unsigned int v4u;
typedef __attribute__((ext_vector_type(16))) __bf16       v16bf;
typedef __attribute__((ext_vector_type(8)))  float        v8f;

#define Bq    4
#define Cc    256
#define Hh    56
#define Wq_   56
#define HW    3136
#define Gg    4
#define NGC   64
#define HEADS 8
#define HC    32
#define Hk    28
#define Wk    28
#define NS    784
#define NSP   800           // NS padded to multiple of 32 (zero filled)
#define SGC   16
#define Lk    28
#define SCALE 0.17677669529663687f  // 32^-0.5
#define EPS   1e-5f

// ---------------- helpers ----------------

__device__ __forceinline__ unsigned short f2bf(float f) {
  union { float f; unsigned int u; } c; c.f = f;
  unsigned int u = c.u;
  u += 0x7FFFu + ((u >> 16) & 1u);     // round-to-nearest-even
  return (unsigned short)(u >> 16);
}
__device__ __forceinline__ float bf2f(unsigned short u) {
  union { unsigned int i; float f; } c; c.i = ((unsigned int)u) << 16;
  return c.f;
}
__device__ __forceinline__ float gelu_exact(float x) {
  return 0.5f * x * (1.0f + erff(x * 0.70710678118654752f));
}

struct __attribute__((aligned(32))) BFPair { v4u lo; v4u hi; };

// Build a 16-element bf16 fragment from two 16-byte chunks (global or LDS).
__device__ __forceinline__ v16bf make_frag(const unsigned short* p0,
                                           const unsigned short* p1) {
  BFPair t;
  t.lo = *(const v4u*)p0;
  t.hi = *(const v4u*)p1;
  v16bf r;
  __builtin_memcpy(&r, &t, sizeof(r));
  return r;
}

__device__ __forceinline__ v8f wmma_bf16(v16bf a, v16bf b, v8f c) {
  return __builtin_amdgcn_wmma_f32_16x16x32_bf16(
      false, a, false, b, (short)0, c, false, false);
}

// ---------------- conversion kernels ----------------

__global__ void cvt_f32_bf16(const float* __restrict__ src,
                             unsigned short* __restrict__ dst, int n) {
  int i = blockIdx.x * blockDim.x + threadIdx.x;
  if (i < n) dst[i] = f2bf(src[i]);
}

// x (B,C,HW) f32 -> x_bf (B*HW, C) bf16   (GEMM A layout)
__global__ void xpose_bf16(const float* __restrict__ x,
                           unsigned short* __restrict__ xbf) {
  int i = blockIdx.x * blockDim.x + threadIdx.x;
  if (i >= Bq * Cc * HW) return;
  int hw = i % HW; int rest = i / HW;
  int c  = rest % Cc; int b = rest / Cc;
  xbf[((size_t)(b * HW + hw)) * Cc + c] = f2bf(x[i]);
}

// v_pc (B*NS, C) f32 -> v_hn (32, HC, NSP) bf16 (zero-padded K)
__global__ void vtrans_bf16(const float* __restrict__ v_pc,
                            unsigned short* __restrict__ v_hn) {
  int i = blockIdx.x * blockDim.x + threadIdx.x;
  if (i >= 32 * HC * NSP) return;
  int nn = i % NSP; int rest = i / NSP;
  int c  = rest % HC; int bh = rest / HC;
  int b = bh >> 3, h = bh & 7;
  float val = 0.0f;
  if (nn < NS) val = v_pc[((size_t)(b * NS + nn)) * Cc + h * HC + c];
  v_hn[i] = f2bf(val);
}

// ---------------- generic bf16 WMMA GEMM (2x2 register blocking) ----------
// C[m,n] = sum_k A[m,k] * Bt[n,k] + bias[n]
// A: M x K row-major (lda), Bt: N x K row-major (ldb).
// Requires K % 32 == 0, M % 32 == 0, N % 32 == 0.
// Each wave computes a 32x32 output block (4 WMMA tiles, fragments reused 2x).
__global__ __launch_bounds__(256) void gemm_bf16(
    const unsigned short* __restrict__ A, const unsigned short* __restrict__ Bt,
    const float* __restrict__ bias, float* __restrict__ Cf,
    unsigned short* __restrict__ Cbf, int M, int N, int K, int lda, int ldb,
    int ldc, int transposeOut, int outHW, int outC) {
  const int tid  = threadIdx.x;
  const int wave = tid >> 5;
  const int lane = tid & 31;
  const int half = lane >> 4;
  const int l16  = lane & 15;
  const int nt = blockIdx.x;                 // 32-wide n block
  const int mt = blockIdx.y * 8 + wave;      // 32-tall m block
  if (mt * 32 >= M) return;  // wave-uniform exit (EXEC stays all-ones)

  const unsigned short* arow0 = A + (size_t)(mt * 32 + l16) * lda;
  const unsigned short* arow1 = arow0 + (size_t)16 * lda;
  const unsigned short* brow0 = Bt + (size_t)(nt * 32 + l16) * ldb;
  const unsigned short* brow1 = brow0 + (size_t)16 * ldb;

  v8f acc00 = {0.f, 0.f, 0.f, 0.f, 0.f, 0.f, 0.f, 0.f};
  v8f acc01 = acc00, acc10 = acc00, acc11 = acc00;
  for (int kk = 0; kk < K; kk += 32) {
    v16bf af0 = make_frag(arow0 + kk + half * 8, arow0 + kk + 16 + half * 8);
    v16bf af1 = make_frag(arow1 + kk + half * 8, arow1 + kk + 16 + half * 8);
    v16bf bf0 = make_frag(brow0 + kk + half * 8, brow0 + kk + 16 + half * 8);
    v16bf bf1 = make_frag(brow1 + kk + half * 8, brow1 + kk + 16 + half * 8);
    acc00 = wmma_bf16(af0, bf0, acc00);
    acc01 = wmma_bf16(af0, bf1, acc01);
    acc10 = wmma_bf16(af1, bf0, acc10);
    acc11 = wmma_bf16(af1, bf1, acc11);
  }

#pragma unroll
  for (int tm = 0; tm < 2; ++tm) {
#pragma unroll
    for (int tn = 0; tn < 2; ++tn) {
      const v8f& acc = tm == 0 ? (tn == 0 ? acc00 : acc01)
                               : (tn == 0 ? acc10 : acc11);
      const int nn = nt * 32 + tn * 16 + l16;
      const float bs = bias ? bias[nn] : 0.0f;
#pragma unroll
      for (int r = 0; r < 8; ++r) {
        int m = mt * 32 + tm * 16 + r + half * 8;
        float v = acc[r] + bs;
        if (Cf) {
          if (transposeOut) {
            int bb = m / outHW, hw = m % outHW;
            Cf[((size_t)(bb * outC + nn)) * outHW + hw] = v;
          } else {
            Cf[(size_t)m * ldc + nn] = v;
          }
        }
        if (Cbf) Cbf[(size_t)m * ldc + nn] = f2bf(v);
      }
    }
  }
}

// ---------------- offset network ----------------

// depthwise 3x3 stride-2 conv on q viewed as (B*G, NGC, 56, 56) -> (BG,NGC,28,28)
__global__ void conv_off(const float* __restrict__ q_phc,
                         const float* __restrict__ w,
                         const float* __restrict__ bvec,
                         float* __restrict__ off_pre) {
  int i = blockIdx.x * blockDim.x + threadIdx.x;
  if (i >= 16 * NGC * Hk * Wk) return;
  int wk = i % Wk; int rest = i / Wk;
  int hk = rest % Hk; rest /= Hk;
  int cg = rest % NGC; int bg = rest / NGC;
  int b = bg >> 2, g = bg & 3;
  int cglob = g * NGC + cg;
  float s = bvec[cg];
#pragma unroll
  for (int ky = 0; ky < 3; ++ky) {
    int hy = hk * 2 + ky - 1;
    if (hy < 0 || hy >= Hh) continue;
#pragma unroll
    for (int kx = 0; kx < 3; ++kx) {
      int wx = wk * 2 + kx - 1;
      if (wx < 0 || wx >= Wq_) continue;
      s += w[cg * 9 + ky * 3 + kx] *
           q_phc[((size_t)(b * HW + hy * Wq_ + wx)) * Cc + cglob];
    }
  }
  off_pre[i] = s;
}

// LN over 64 channels + exact GELU + pointwise(2) -> off_xy (BG*784, 2)
__global__ void off_ln_pw(const float* __restrict__ off_pre,
                          const float* __restrict__ g,
                          const float* __restrict__ bln,
                          const float* __restrict__ pw,
                          float* __restrict__ off_xy) {
  int t = blockIdx.x * blockDim.x + threadIdx.x;
  if (t >= 16 * NS) return;
  int bg = t / NS, sp = t % NS;
  const float* base = off_pre + (size_t)(bg * NGC) * NS + sp;
  float sum = 0.f, sq = 0.f;
  for (int c = 0; c < NGC; ++c) { float v = base[c * NS]; sum += v; sq += v * v; }
  float mean = sum / NGC;
  float var  = sq / NGC - mean * mean;
  float rstd = rsqrtf(var + EPS);
  float s0 = 0.f, s1 = 0.f;
  for (int c = 0; c < NGC; ++c) {
    float xn = (base[c * NS] - mean) * rstd * g[c] + bln[c];
    float ge = gelu_exact(xn);
    s0 += pw[c] * ge;
    s1 += pw[NGC + c] * ge;
  }
  off_xy[t * 2 + 0] = s0;
  off_xy[t * 2 + 1] = s1;
}

// depthwise 1D k=3 stride-2 on Str_Zab viewed (16, SGC, 56) -> (16,SGC,28)
__global__ void ab_dw(const float* __restrict__ sz, const float* __restrict__ w,
                      const float* __restrict__ bvec, float* __restrict__ ab_pre) {
  int i = blockIdx.x * blockDim.x + threadIdx.x;
  if (i >= 16 * SGC * Lk) return;
  int l = i % Lk; int rest = i / Lk;
  int cg = rest % SGC; int bg = rest / SGC;
  float s = bvec[cg];
#pragma unroll
  for (int k = 0; k < 3; ++k) {
    int p = l * 2 + k - 1;
    if (p < 0 || p >= Wq_) continue;
    s += w[cg * 3 + k] * sz[(size_t)(bg * SGC + cg) * Wq_ + p];
  }
  ab_pre[i] = s;
}

__global__ void ab_ln_pw(const float* __restrict__ ab_pre,
                         const float* __restrict__ g,
                         const float* __restrict__ bln,
                         const float* __restrict__ pw,
                         float* __restrict__ ab_out) {
  int t = blockIdx.x * blockDim.x + threadIdx.x;
  if (t >= 16 * Lk) return;
  int bg = t / Lk, l = t % Lk;
  const float* base = ab_pre + (size_t)(bg * SGC) * Lk + l;
  float sum = 0.f, sq = 0.f;
  for (int c = 0; c < SGC; ++c) { float v = base[c * Lk]; sum += v; sq += v * v; }
  float mean = sum / SGC;
  float var  = sq / SGC - mean * mean;
  float rstd = rsqrtf(var + EPS);
  float s0 = 0.f, s1 = 0.f;
  for (int c = 0; c < SGC; ++c) {
    float xn = (base[c * Lk] - mean) * rstd * g[c] + bln[c];
    float ge = gelu_exact(xn);
    s0 += pw[c] * ge;
    s1 += pw[SGC + c] * ge;
  }
  ab_out[t * 2 + 0] = s0;
  ab_out[t * 2 + 1] = s1;
}

// ---------------- deformable sampling ----------------
// bilinear grid_sample -> xs_bf (B*NS, C) bf16  (GEMM-A layout for k/v)
__global__ void grid_sample_k(const float* __restrict__ x,
                              const float* __restrict__ off_xy,
                              const float* __restrict__ ab,
                              unsigned short* __restrict__ xs_bf) {
  int idx = blockIdx.x * blockDim.x + threadIdx.x;
  if (idx >= 16 * NS) return;
  int bg = idx / NS, j = idx % NS;
  int hk = j / Wk, wk = j % Wk;
  int b = bg >> 2, g = bg & 3;

  float ry = (0.5f + (float)hk) / 27.0f * 2.0f - 1.0f;
  float rx = (0.5f + (float)wk) / 27.0f * 2.0f - 1.0f;
  float py = off_xy[idx * 2 + 0] + ab[(bg * Lk + wk) * 2 + 0] + ry;
  float px = off_xy[idx * 2 + 1] + ab[(bg * Lk + wk) * 2 + 1] + rx;
  py = fminf(1.f, fmaxf(-1.f, py));
  px = fminf(1.f, fmaxf(-1.f, px));

  float gx = (px + 1.f) * 0.5f * (float)(Wq_ - 1);
  float gy = (py + 1.f) * 0.5f * (float)(Hh - 1);
  float x0f = floorf(gx), y0f = floorf(gy);
  float wx = gx - x0f, wy = gy - y0f;
  int x0i = (int)x0f, y0i = (int)y0f;
  int x1i = x0i + 1, y1i = y0i + 1;
  float vx0 = (x0i >= 0 && x0i < Wq_) ? 1.f : 0.f;
  float vx1 = (x1i >= 0 && x1i < Wq_) ? 1.f : 0.f;
  float vy0 = (y0i >= 0 && y0i < Hh) ? 1.f : 0.f;
  float vy1 = (y1i >= 0 && y1i < Hh) ? 1.f : 0.f;
  int x0c = min(max(x0i, 0), Wq_ - 1), x1c = min(max(x1i, 0), Wq_ - 1);
  int y0c = min(max(y0i, 0), Hh - 1),  y1c = min(max(y1i, 0), Hh - 1);
  float w00 = vy0 * vx0 * (1.f - wx) * (1.f - wy);
  float w01 = vy0 * vx1 * wx * (1.f - wy);
  float w10 = vy1 * vx0 * (1.f - wx) * wy;
  float w11 = vy1 * vx1 * wx * wy;

  for (int cg = 0; cg < NGC; ++cg) {
    const float* img = x + ((size_t)(b * Cc + g * NGC + cg)) * (Hh * Wq_);
    float val = img[y0c * Wq_ + x0c] * w00 + img[y0c * Wq_ + x1c] * w01 +
                img[y1c * Wq_ + x0c] * w10 + img[y1c * Wq_ + x1c] * w11;
    xs_bf[((size_t)(b * NS + j)) * Cc + g * NGC + cg] = f2bf(val);
  }
}

// ---------------- LEPE (3x3 depthwise, stride 1) ----------------
__global__ void lepe_k(const float* __restrict__ q_phc,
                       const float* __restrict__ w,
                       const float* __restrict__ bvec,
                       float* __restrict__ lepe) {
  int i = blockIdx.x * blockDim.x + threadIdx.x;
  if (i >= Bq * HW * Cc) return;
  int c = i % Cc; int rest = i / Cc;
  int hw = rest % HW; int b = rest / HW;
  int h = hw / Wq_, ww = hw % Wq_;
  float s = bvec[c];
#pragma unroll
  for (int ky = 0; ky < 3; ++ky) {
    int y = h + ky - 1;
    if (y < 0 || y >= Hh) continue;
#pragma unroll
    for (int kx = 0; kx < 3; ++kx) {
      int xq = ww + kx - 1;
      if (xq < 0 || xq >= Wq_) continue;
      s += w[c * 9 + ky * 3 + kx] *
           q_phc[((size_t)(b * HW + y * Wq_ + xq)) * Cc + c];
    }
  }
  lepe[i] = s;
}

// ---------------- fused attention ----------------
// grid (HW/16, 32). Per block: 16 query rows of one (b,h).
// Phase 1: logits via WMMA (K=HC=32) -> bf16 LDS [16][NSP]
// Phase 2: row softmax (8 threads/row, LDS reduction, in-place bf16)
// Phase 3: probs @ V via WMMA (K=NSP), + LEPE, emit bf16 for o-proj.
__global__ __launch_bounds__(128) void attn_fused(
    const unsigned short* __restrict__ qbf, const unsigned short* __restrict__ kbf,
    const unsigned short* __restrict__ vhn, const float* __restrict__ lepe,
    unsigned short* __restrict__ outbf) {
  __shared__ unsigned short probs[16][NSP];
  __shared__ float red[16][8];

  const int mt = blockIdx.x;
  const int bh = blockIdx.y;
  const int b = bh >> 3, h = bh & 7;
  const int tid  = threadIdx.x;
  const int wave = tid >> 5;
  const int lane = tid & 31;
  const int half = lane >> 4;
  const int l16  = lane & 15;
  const int m0 = mt * 16;

  // zero the K-pad columns
  for (int i = tid; i < 16 * (NSP - NS); i += 128)
    probs[i / (NSP - NS)][NS + i % (NSP - NS)] = 0;

  // ---- phase 1: logits ----
  const unsigned short* qrow =
      qbf + ((size_t)(b * HW + m0 + l16)) * Cc + h * HC;
  v16bf afrag = make_frag(qrow + half * 8, qrow + 16 + half * 8);
  for (int nt = wave; nt < NS / 16; nt += 4) {
    const unsigned short* krow =
        kbf + ((size_t)(b * NS + nt * 16 + l16)) * Cc + h * HC;
    v16bf bfrag = make_frag(krow + half * 8, krow + 16 + half * 8);
    v8f acc = {0.f, 0.f, 0.f, 0.f, 0.f, 0.f, 0.f, 0.f};
    acc = wmma_bf16(afrag, bfrag, acc);
#pragma unroll
    for (int r = 0; r < 8; ++r)
      probs[r + half * 8][nt * 16 + l16] = f2bf(acc[r] * SCALE);
  }
  __syncthreads();

  // ---- phase 2: softmax, 8 threads per row ----
  {
    const int row = tid >> 3;
    const int sub = tid & 7;
    float mx = -1e30f;
    for (int j = sub; j < NS; j += 8) mx = fmaxf(mx, bf2f(probs[row][j]));
    red[row][sub] = mx;
    __syncthreads();
    float m = -1e30f;
#pragma unroll
    for (int k = 0; k < 8; ++k) m = fmaxf(m, red[row][k]);
    __syncthreads();
    float s = 0.f;
    for (int j = sub; j < NS; j += 8) {
      float e = __expf(bf2f(probs[row][j]) - m);
      s += e;
      probs[row][j] = f2bf(e);
    }
    red[row][sub] = s;
    __syncthreads();
    float tot = 0.f;
#pragma unroll
    for (int k = 0; k < 8; ++k) tot += red[row][k];
    float inv = 1.0f / tot;
    for (int j = sub; j < NS; j += 8)
      probs[row][j] = f2bf(bf2f(probs[row][j]) * inv);
  }
  __syncthreads();

  // ---- phase 3: probs @ V  (waves 0,1 each handle 16 of HC=32 channels) ----
  if (wave < 2) {
    const int ct = wave;
    const unsigned short* vrow =
        vhn + ((size_t)(bh * HC + ct * 16 + l16)) * NSP;
    v8f acc = {0.f, 0.f, 0.f, 0.f, 0.f, 0.f, 0.f, 0.f};
    for (int kk = 0; kk < NSP; kk += 32) {
      v16bf af = make_frag(&probs[l16][kk + half * 8],
                           &probs[l16][kk + 16 + half * 8]);
      v16bf bf = make_frag(vrow + kk + half * 8, vrow + kk + 16 + half * 8);
      acc = wmma_bf16(af, bf, acc);
    }
#pragma unroll
    for (int r = 0; r < 8; ++r) {
      int m = m0 + r + half * 8;
      int cg = h * HC + ct * 16 + l16;
      size_t idx = ((size_t)(b * HW + m)) * Cc + cg;
      outbf[idx] = f2bf(acc[r] + lepe[idx]);
    }
  }
}

// ---------------- host side ----------------

extern "C" void kernel_launch(void* const* d_in, const int* in_sizes, int n_in,
                              void* d_out, int out_size, void* d_ws,
                              size_t ws_size, hipStream_t stream) {
  const float* x      = (const float*)d_in[0];
  const float* strz   = (const float*)d_in[1];
  const float* codww  = (const float*)d_in[2];
  const float* codwb  = (const float*)d_in[3];
  const float* colng  = (const float*)d_in[4];
  const float* colnb  = (const float*)d_in[5];
  const float* copw   = (const float*)d_in[6];
  const float* abdww  = (const float*)d_in[7];
  const float* abdwb  = (const float*)d_in[8];
  const float* ablng  = (const float*)d_in[9];
  const float* ablnb  = (const float*)d_in[10];
  const float* abpw   = (const float*)d_in[11];
  const float* wq     = (const float*)d_in[12];
  const float* bq     = (const float*)d_in[13];
  const float* wk     = (const float*)d_in[14];
  const float* bk     = (const float*)d_in[15];
  const float* wv     = (const float*)d_in[16];
  const float* bv     = (const float*)d_in[17];
  const float* wo     = (const float*)d_in[18];
  const float* bo     = (const float*)d_in[19];
  const float* rpew   = (const float*)d_in[20];
  const float* rpeb   = (const float*)d_in[21];
  float* out = (float*)d_out;

  char* wsp = (char*)d_ws;
  auto alloc = [&](size_t bytes) -> void* {
    void* p = (void*)wsp;
    wsp += (bytes + 255) & ~(size_t)255;
    return p;
  };

  const size_t PC = (size_t)Bq * HW * Cc;   // 12544*256
  const size_t SC = (size_t)Bq * NS * Cc;   //  3136*256

  unsigned short* wq_bf  = (unsigned short*)alloc(65536 * 2);
  unsigned short* wk_bf  = (unsigned short*)alloc(65536 * 2);
  unsigned short* wv_bf  = (unsigned short*)alloc(65536 * 2);
  unsigned short* wo_bf  = (unsigned short*)alloc(65536 * 2);
  unsigned short* x_bf   = (unsigned short*)alloc(PC * 2);
  float*          q_phc  = (float*)alloc(PC * 4);
  unsigned short* q_bf   = (unsigned short*)alloc(PC * 2);
  float*          offpre = (float*)alloc((size_t)16 * NGC * NS * 4);
  float*          offxy  = (float*)alloc((size_t)16 * NS * 2 * 4);
  float*          abpre  = (float*)alloc((size_t)16 * SGC * Lk * 4);
  float*          abv    = (float*)alloc((size_t)16 * Lk * 2 * 4);
  unsigned short* xs_bf  = (unsigned short*)alloc(SC * 2);
  unsigned short* k_bf   = (unsigned short*)alloc(SC * 2);
  float*          v_pc   = (float*)alloc(SC * 4);
  unsigned short* v_hn   = (unsigned short*)alloc((size_t)32 * HC * NSP * 2);
  float*          lepe   = (float*)alloc(PC * 4);
  unsigned short* out_bf = (unsigned short*)alloc(PC * 2);

  const int TB = 256;
  // weights -> bf16
  cvt_f32_bf16<<<(65536 + TB - 1) / TB, TB, 0, stream>>>(wq, wq_bf, 65536);
  cvt_f32_bf16<<<(65536 + TB - 1) / TB, TB, 0, stream>>>(wk, wk_bf, 65536);
  cvt_f32_bf16<<<(65536 + TB - 1) / TB, TB, 0, stream>>>(wv, wv_bf, 65536);
  cvt_f32_bf16<<<(65536 + TB - 1) / TB, TB, 0, stream>>>(wo, wo_bf, 65536);
  // x -> (pixels, C) bf16
  xpose_bf16<<<(int)((PC + TB - 1) / TB), TB, 0, stream>>>(x, x_bf);

  // q = x @ wq^T + bq  (M=12544, N=K=256); f32 + bf16 outputs
  {
    int M = (int)(PC / Cc);
    gemm_bf16<<<dim3(Cc / 32, (M / 32 + 7) / 8), TB, 0, stream>>>(
        x_bf, wq_bf, bq, q_phc, q_bf, M, Cc, Cc, Cc, Cc, Cc, 0, 0, 0);
  }

  // offset network
  conv_off<<<(16 * NGC * NS + TB - 1) / TB, TB, 0, stream>>>(q_phc, codww,
                                                             codwb, offpre);
  off_ln_pw<<<(16 * NS + TB - 1) / TB, TB, 0, stream>>>(offpre, colng, colnb,
                                                        copw, offxy);
  ab_dw<<<(16 * SGC * Lk + TB - 1) / TB, TB, 0, stream>>>(strz, abdww, abdwb,
                                                          abpre);
  ab_ln_pw<<<(16 * Lk + TB - 1) / TB, TB, 0, stream>>>(abpre, ablng, ablnb,
                                                       abpw, abv);
  // deformable sampling -> xs (bf16, GEMM-A layout)
  grid_sample_k<<<(16 * NS + TB - 1) / TB, TB, 0, stream>>>(x, offxy, abv,
                                                            xs_bf);

  // k = xs @ wk^T + bk (bf16 out);  v = xs @ wv^T + bv (f32 out)
  {
    int M = Bq * NS;  // 3136
    gemm_bf16<<<dim3(Cc / 32, (M / 32 + 7) / 8), TB, 0, stream>>>(
        xs_bf, wk_bf, bk, (float*)nullptr, k_bf, M, Cc, Cc, Cc, Cc, Cc, 0, 0,
        0);
    gemm_bf16<<<dim3(Cc / 32, (M / 32 + 7) / 8), TB, 0, stream>>>(
        xs_bf, wv_bf, bv, v_pc, (unsigned short*)nullptr, M, Cc, Cc, Cc, Cc,
        Cc, 0, 0, 0);
  }
  // v -> (bh, HC, NSP) bf16 (zero padded)
  vtrans_bf16<<<(32 * HC * NSP + TB - 1) / TB, TB, 0, stream>>>(v_pc, v_hn);

  // LEPE
  lepe_k<<<(int)((PC + TB - 1) / TB), TB, 0, stream>>>(q_phc, rpew, rpeb, lepe);

  // fused attention (+LEPE) -> out_bf (pixels, C)
  attn_fused<<<dim3(HW / 16, Bq * HEADS), 128, 0, stream>>>(q_bf, k_bf, v_hn,
                                                            lepe, out_bf);

  // final o-projection, transposed store straight into d_out as (B,C,H,W)
  {
    int M = (int)(PC / Cc);
    gemm_bf16<<<dim3(Cc / 32, (M / 32 + 7) / 8), TB, 0, stream>>>(
        out_bf, wo_bf, bo, out, (unsigned short*)nullptr, M, Cc, Cc, Cc, Cc,
        Cc, 1, HW, Cc);
  }
}
